// TopPSkippedQwen3MoeSparseMoeBlock_72756745994645
// MI455X (gfx1250) — matrix operand order repaired and checked by reading, not compiled, not run on hardware
//
#include <hip/hip_runtime.h>
#include <hip/hip_bf16.h>
#include <math.h>

// ---------------------------------------------------------------------------
// Types for CDNA5 WMMA (wave32): A=16x32 bf16, B=32x16 bf16, C/D=16x16 f32
// ---------------------------------------------------------------------------
typedef __attribute__((ext_vector_type(16))) __bf16 v16bf;
typedef __attribute__((ext_vector_type(8)))  __bf16 v8bf;
typedef __attribute__((ext_vector_type(8)))  float  v8f;

#define T_TOKENS 2048
#define HIDDEN   2048
#define INTER    768
#define NEXP     32
#define TOPK     8
#define THRESH   0.8f
#define MAXSLOTS (T_TOKENS * TOPK)

// ---------------------------------------------------------------------------
// K0: zero output accumulator and per-expert counts
// ---------------------------------------------------------------------------
__global__ void moe_init(float* __restrict__ out, int* __restrict__ counts, int n)
{
    int i = blockIdx.x * blockDim.x + threadIdx.x;
    for (; i < n; i += gridDim.x * blockDim.x) out[i] = 0.0f;
    if (blockIdx.x == 0 && threadIdx.x < NEXP) counts[threadIdx.x] = 0;
}

// ---------------------------------------------------------------------------
// K1: router. One wave (32 lanes) per token; lane e owns expert e.
// ---------------------------------------------------------------------------
__global__ __launch_bounds__(32) void moe_router(
    const float* __restrict__ x, const float* __restrict__ gw,
    float* __restrict__ combine, int* __restrict__ counts)
{
    const int t    = blockIdx.x;
    const int lane = threadIdx.x;

    __shared__ float xs[HIDDEN];
    for (int i = lane; i < HIDDEN; i += 32) xs[i] = x[(size_t)t * HIDDEN + i];
    __syncthreads();

    const float* w = gw + (size_t)lane * HIDDEN;
    float acc = 0.0f;
    for (int h = 0; h < HIDDEN; ++h) acc = fmaf(xs[h], w[h], acc);

    float mx = acc;
    for (int off = 16; off >= 1; off >>= 1) mx = fmaxf(mx, __shfl_xor(mx, off, 32));
    float ex  = __expf(acc - mx);
    float sum = ex;
    for (int off = 16; off >= 1; off >>= 1) sum += __shfl_xor(sum, off, 32);
    const float p = ex / sum;

    bool  sel = false;
    float topv[TOPK];
    int   topi[TOPK];
    for (int k = 0; k < TOPK; ++k) {
        float v   = sel ? -1.0f : p;
        int   idx = lane;
        for (int off = 16; off >= 1; off >>= 1) {
            float ov = __shfl_xor(v, off, 32);
            int   oi = __shfl_xor(idx, off, 32);
            if (ov > v || (ov == v && oi < idx)) { v = ov; idx = oi; }
        }
        topv[k] = v;
        topi[k] = idx;
        if (lane == idx) sel = true;
    }

    float s8 = 0.0f;
    for (int k = 0; k < TOPK; ++k) s8 += topv[k];
    const float inv = 1.0f / fmaxf(s8, 1e-12f);

    float cum = 0.0f; int cnt = 0;
    for (int k = 0; k < TOPK; ++k) {
        cum += topv[k] * inv;
        if (cum < THRESH) cnt++;
    }
    const int nk = min(cnt + 1, TOPK);

    float ks = 0.0f;
    for (int k = 0; k < nk; ++k) ks += topv[k];
    const float kinv = 1.0f / fmaxf(ks, 1e-12f);

    float myw = 0.0f;
    for (int k = 0; k < nk; ++k)
        if (topi[k] == lane) myw = topv[k] * kinv;

    combine[(size_t)t * NEXP + lane] = myw;
    if (myw > 0.0f) atomicAdd(&counts[lane], 1);
}

// ---------------------------------------------------------------------------
// K2: scan over 32 expert counts (single wave)
// ---------------------------------------------------------------------------
__global__ __launch_bounds__(32) void moe_scan(
    const int* __restrict__ counts, int* __restrict__ offsets, int* __restrict__ cursors)
{
    const int lane = threadIdx.x;
    const int c = counts[lane];
    int v = c;
    for (int off = 1; off < 32; off <<= 1) {
        int o = __shfl_up(v, off, 32);
        if (lane >= off) v += o;
    }
    offsets[lane + 1] = v;
    if (lane == 0) offsets[0] = 0;
    cursors[lane] = v - c;
}

// ---------------------------------------------------------------------------
// K3: scatter (token, weight) into per-expert slot segments
// ---------------------------------------------------------------------------
__global__ void moe_scatter(
    const float* __restrict__ combine, int* __restrict__ cursors,
    int* __restrict__ token_list, float* __restrict__ slot_w)
{
    const int t = blockIdx.x * blockDim.x + threadIdx.x;
    if (t >= T_TOKENS) return;
    for (int e = 0; e < NEXP; ++e) {
        float w = combine[(size_t)t * NEXP + e];
        if (w > 0.0f) {
            int pos = atomicAdd(&cursors[e], 1);
            token_list[pos] = t;
            slot_w[pos]     = w;
        }
    }
}

// ---------------------------------------------------------------------------
// K3b: vectorized fp32 -> bf16 conversion (8 elements / thread / iter)
// ---------------------------------------------------------------------------
__global__ void moe_convert8(const float* __restrict__ in, unsigned short* __restrict__ outp, int n8)
{
    __bf16* o = reinterpret_cast<__bf16*>(outp);
    int i = blockIdx.x * blockDim.x + threadIdx.x;
    for (; i < n8; i += gridDim.x * blockDim.x) {
        const float4* p = (const float4*)(in + (size_t)i * 8);
        float4 f0 = p[0], f1 = p[1];
        v8bf v;
        v[0] = (__bf16)f0.x; v[1] = (__bf16)f0.y; v[2] = (__bf16)f0.z; v[3] = (__bf16)f0.w;
        v[4] = (__bf16)f1.x; v[5] = (__bf16)f1.y; v[6] = (__bf16)f1.z; v[7] = (__bf16)f1.w;
        *(v8bf*)(o + (size_t)i * 8) = v;
    }
}

// ---------------------------------------------------------------------------
// Fragment loaders
// ---------------------------------------------------------------------------
template<bool WB16>
__device__ __forceinline__ v16bf load_wfrag(const float* wf, const __bf16* wb)
{
    if constexpr (WB16) {
        return *(const v16bf*)wb;  // 16 consecutive bf16, 32B aligned
    } else {
        v16bf b;
        const float4* p = (const float4*)wf;
#pragma unroll
        for (int q = 0; q < 4; ++q) {
            float4 f = p[q];
            b[q*4+0] = (__bf16)f.x; b[q*4+1] = (__bf16)f.y;
            b[q*4+2] = (__bf16)f.z; b[q*4+3] = (__bf16)f.w;
        }
        return b;
    }
}

__device__ __forceinline__ v16bf load_afrag(const __bf16* xrow, int kc, int aoff)
{
    const __bf16* xp = xrow + kc * 32 + aoff;
    v8bf alo = *(const v8bf*)xp;
    v8bf ahi = *(const v8bf*)(xp + 8);
    return __builtin_shufflevector(alo, ahi, 0,1,2,3,4,5,6,7,8,9,10,11,12,13,14,15);
}

// ---------------------------------------------------------------------------
// K4: gate_up GEMM + SwiGLU. 512 threads = 16 waves; wave owns 3 gate/up
// tile pairs. Software-pipelined double buffering: loads of chunk k+1 are
// issued before the WMMAs of chunk k -> partial s_wait_loadcnt, real overlap.
// ---------------------------------------------------------------------------
struct FragsUp { v16bf a; v16bf bg[3]; v16bf bu[3]; };

template<bool WB16>
__device__ __forceinline__ void load_chunk_up(
    FragsUp& f, const __bf16* xrow, int kc, int aoff,
    const float* wf, const __bf16* wb, size_t UOFF)
{
    f.a = load_afrag(xrow, kc, aoff);
#pragma unroll
    for (int j = 0; j < 3; ++j) {
        const size_t off = (size_t)j * 16 * HIDDEN + kc * 32;
        f.bg[j] = load_wfrag<WB16>(wf + off,        wb + off);
        f.bu[j] = load_wfrag<WB16>(wf + off + UOFF, wb + off + UOFF);
    }
}

__device__ __forceinline__ void mma_chunk_up(const FragsUp& f, v8f accG[3], v8f accU[3])
{
#pragma unroll
    for (int j = 0; j < 3; ++j) {
        accG[j] = __builtin_amdgcn_wmma_f32_16x16x32_bf16(
            false, f.a, false, f.bg[j], (short)0, accG[j], false, false);
        accU[j] = __builtin_amdgcn_wmma_f32_16x16x32_bf16(
            false, f.a, false, f.bu[j], (short)0, accU[j], false, false);
    }
}

template<bool WB16>
__global__ __launch_bounds__(512) void moe_gemm_up(
    const unsigned short* __restrict__ xb_, const float* __restrict__ wguf,
    const unsigned short* __restrict__ wgub_,
    const int* __restrict__ token_list, const int* __restrict__ offsets,
    unsigned short* __restrict__ hbuf_)
{
    const __bf16* xb   = reinterpret_cast<const __bf16*>(xb_);
    __bf16*       hbuf = reinterpret_cast<__bf16*>(hbuf_);

    const int e    = blockIdx.y;
    const int seg0 = offsets[e];
    const int seg1 = offsets[e + 1];
    const int base = seg0 + blockIdx.x * 16;
    if (base >= seg1) return;

    const int lane = threadIdx.x & 31;
    const int wave = threadIdx.x >> 5;     // 0..15
    const int row  = lane & 15;
    const int hi   = lane >> 4;
    const int aoff = hi * 16;
    const int koff = hi * 16;
    const int ncol = lane & 15;

    const int slot = base + row;
    const int tok  = token_list[slot < seg1 ? slot : seg0];
    const __bf16* xrow = xb + (size_t)tok * HIDDEN;

    const size_t wbase = ((size_t)e * (2 * INTER) + wave * 48 + ncol) * HIDDEN + koff;
    const float*  wf = wguf + wbase;
    const __bf16* wb = reinterpret_cast<const __bf16*>(wgub_) + wbase;
    const size_t UOFF = (size_t)INTER * HIDDEN;

    v8f accG[3], accU[3];
#pragma unroll
    for (int j = 0; j < 3; ++j) {
        accG[j] = (v8f){0,0,0,0,0,0,0,0};
        accU[j] = (v8f){0,0,0,0,0,0,0,0};
    }

    const int NKC = HIDDEN / 32;           // 64 chunks
    FragsUp f0, f1;
    load_chunk_up<WB16>(f0, xrow, 0, aoff, wf, wb, UOFF);
    for (int kc = 0; kc < NKC - 2; kc += 2) {
        load_chunk_up<WB16>(f1, xrow, kc + 1, aoff, wf, wb, UOFF);
        mma_chunk_up(f0, accG, accU);
        load_chunk_up<WB16>(f0, xrow, kc + 2, aoff, wf, wb, UOFF);
        mma_chunk_up(f1, accG, accU);
    }
    load_chunk_up<WB16>(f1, xrow, NKC - 1, aoff, wf, wb, UOFF);
    mma_chunk_up(f0, accG, accU);
    mma_chunk_up(f1, accG, accU);

    // epilogue: h = silu(g)*u ; C layout: VGPR r -> M = r + 8*hi, N = lane&15
    if (base + 16 <= seg1) {  // full tile: wave-uniform fast path
#pragma unroll
        for (int j = 0; j < 3; ++j) {
#pragma unroll
            for (int r = 0; r < 8; ++r) {
                const int s = base + r + hi * 8;
                float g  = accG[j][r];
                float u  = accU[j][r];
                float hv = g * __builtin_amdgcn_rcpf(1.0f + __expf(-g)) * u;
                hbuf[(size_t)s * INTER + wave * 48 + j * 16 + ncol] = (__bf16)hv;
            }
        }
    } else {                  // tail tile
#pragma unroll
        for (int j = 0; j < 3; ++j) {
#pragma unroll
            for (int r = 0; r < 8; ++r) {
                const int s = base + r + hi * 8;
                if (s < seg1) {
                    float g  = accG[j][r];
                    float u  = accU[j][r];
                    float hv = g * __builtin_amdgcn_rcpf(1.0f + __expf(-g)) * u;
                    hbuf[(size_t)s * INTER + wave * 48 + j * 16 + ncol] = (__bf16)hv;
                }
            }
        }
    }
}

// ---------------------------------------------------------------------------
// K5: down GEMM + weighted scatter-accumulate. 1024 threads = 32 waves; wave
// owns 4 n-tiles (64 output cols). Same double-buffered pipeline as K4.
// Tail rows clamped to a valid slot with weight 0.0 -> epilogue unguarded.
// ---------------------------------------------------------------------------
struct FragsDn { v16bf a; v16bf b[4]; };

template<bool WB16>
__device__ __forceinline__ void load_chunk_dn(
    FragsDn& f, const __bf16* hrow, int kc, int aoff,
    const float* wf, const __bf16* wb)
{
    f.a = load_afrag(hrow, kc, aoff);
#pragma unroll
    for (int j = 0; j < 4; ++j) {
        const size_t off = (size_t)j * 16 * INTER + kc * 32;
        f.b[j] = load_wfrag<WB16>(wf + off, wb + off);
    }
}

__device__ __forceinline__ void mma_chunk_dn(const FragsDn& f, v8f acc[4])
{
#pragma unroll
    for (int j = 0; j < 4; ++j) {
        acc[j] = __builtin_amdgcn_wmma_f32_16x16x32_bf16(
            false, f.a, false, f.b[j], (short)0, acc[j], false, false);
    }
}

template<bool WB16>
__global__ __launch_bounds__(1024) void moe_gemm_down(
    const unsigned short* __restrict__ hbuf_, const float* __restrict__ wdf,
    const unsigned short* __restrict__ wdb_,
    const int* __restrict__ token_list, const float* __restrict__ slot_w,
    const int* __restrict__ offsets, float* __restrict__ out)
{
    const __bf16* hbuf = reinterpret_cast<const __bf16*>(hbuf_);

    const int e    = blockIdx.y;
    const int seg0 = offsets[e];
    const int seg1 = offsets[e + 1];
    const int base = seg0 + blockIdx.x * 16;
    if (base >= seg1) return;

    const int lane = threadIdx.x & 31;
    const int wave = threadIdx.x >> 5;     // 0..31
    const int row  = lane & 15;
    const int hi   = lane >> 4;
    const int aoff = hi * 16;
    const int koff = hi * 16;
    const int ncol = lane & 15;

    const int slot = base + row;
    const int sc   = (slot < seg1) ? slot : seg0;
    const __bf16* hrow = hbuf + (size_t)sc * INTER;

    int   toks[8];
    float wgt[8];
#pragma unroll
    for (int r = 0; r < 8; ++r) {
        const int s  = base + r + hi * 8;
        const int s2 = (s < seg1) ? s : seg0;
        toks[r] = token_list[s2];
        float wv = slot_w[s2];
        wgt[r]  = (s < seg1) ? wv : 0.0f;
    }

    // wave covers output columns [wave*64, wave*64+64)
    const size_t wbase = ((size_t)e * HIDDEN + wave * 64 + ncol) * INTER + koff;
    const float*  wf = wdf + wbase;
    const __bf16* wb = reinterpret_cast<const __bf16*>(wdb_) + wbase;

    v8f acc[4];
#pragma unroll
    for (int j = 0; j < 4; ++j) acc[j] = (v8f){0,0,0,0,0,0,0,0};

    const int NKC = INTER / 32;            // 24 chunks
    FragsDn f0, f1;
    load_chunk_dn<WB16>(f0, hrow, 0, aoff, wf, wb);
    for (int kc = 0; kc < NKC - 2; kc += 2) {
        load_chunk_dn<WB16>(f1, hrow, kc + 1, aoff, wf, wb);
        mma_chunk_dn(f0, acc);
        load_chunk_dn<WB16>(f0, hrow, kc + 2, aoff, wf, wb);
        mma_chunk_dn(f1, acc);
    }
    load_chunk_dn<WB16>(f1, hrow, NKC - 1, aoff, wf, wb);
    mma_chunk_dn(f0, acc);
    mma_chunk_dn(f1, acc);

#pragma unroll
    for (int j = 0; j < 4; ++j) {
#pragma unroll
        for (int r = 0; r < 8; ++r) {
            atomicAdd(&out[(size_t)toks[r] * HIDDEN + wave * 64 + j * 16 + ncol],
                      acc[j][r] * wgt[r]);
        }
    }
}

// ---------------------------------------------------------------------------
// Host launcher
// ---------------------------------------------------------------------------
extern "C" void kernel_launch(void* const* d_in, const int* in_sizes, int n_in,
                              void* d_out, int out_size, void* d_ws, size_t ws_size,
                              hipStream_t stream) {
    const float* x   = (const float*)d_in[0];  // [2,1024,2048]
    const float* gw  = (const float*)d_in[1];  // [32,2048]
    const float* wgu = (const float*)d_in[2];  // [32,1536,2048]
    const float* wd  = (const float*)d_in[3];  // [32,2048,768]
    float* out = (float*)d_out;                // [2,1024,2048]

    char* ws = (char*)d_ws;
    float*          combine    = (float*)(ws + 0);          // 262144
    int*            counts     = (int*)(ws + 262144);
    int*            offsets    = (int*)(ws + 262400);
    int*            cursors    = (int*)(ws + 262656);
    int*            token_list = (int*)(ws + 262912);       // 65536
    float*          slot_w     = (float*)(ws + 328448);     // 65536
    unsigned short* xb         = (unsigned short*)(ws + 393984);           // 8 MB
    unsigned short* hbuf       = (unsigned short*)(ws + 393984 + 8388608); // 24 MB
    const size_t    BASE_END   = 393984ull + 8388608ull + 25165824ull;     // ~34 MB

    const int NWGU = NEXP * 2 * INTER * HIDDEN;   // 100,663,296
    const int NWD  = NEXP * HIDDEN * INTER;       //  50,331,648
    unsigned short* wgub = (unsigned short*)(ws + BASE_END);
    unsigned short* wdb  = (unsigned short*)(ws + BASE_END + (size_t)NWGU * 2);
    const size_t BF16_END = BASE_END + (size_t)NWGU * 2 + (size_t)NWD * 2;  // ~336 MB
    const bool wb16 = (ws_size >= BF16_END);

    const int nout = T_TOKENS * HIDDEN;

    moe_init<<<1024, 256, 0, stream>>>(out, counts, nout);
    moe_router<<<T_TOKENS, 32, 0, stream>>>(x, gw, combine, counts);
    moe_scan<<<1, 32, 0, stream>>>(counts, offsets, cursors);
    moe_scatter<<<(T_TOKENS + 255) / 256, 256, 0, stream>>>(combine, cursors, token_list, slot_w);
    moe_convert8<<<2048, 256, 0, stream>>>(x, xb, (T_TOKENS * HIDDEN) / 8);

    dim3 ggrid(T_TOKENS / 16, NEXP);
    if (wb16) {
        moe_convert8<<<8192, 256, 0, stream>>>(wgu, wgub, NWGU / 8);
        moe_convert8<<<8192, 256, 0, stream>>>(wd,  wdb,  NWD  / 8);
        moe_gemm_up<true><<<ggrid, 512, 0, stream>>>(xb, wgu, wgub, token_list, offsets, hbuf);
        moe_gemm_down<true><<<ggrid, 1024, 0, stream>>>(hbuf, wd, wdb, token_list, slot_w, offsets, out);
    } else {
        moe_gemm_up<false><<<ggrid, 512, 0, stream>>>(xb, wgu, wgub, token_list, offsets, hbuf);
        moe_gemm_down<false><<<ggrid, 1024, 0, stream>>>(hbuf, wd, wdb, token_list, slot_w, offsets, out);
    }
}
